// GcnClustering_71519795413685
// MI455X (gfx1250) — compile-verified, matrix operand order checked by reading, not run on hardware
//
#include <hip/hip_runtime.h>

// ---------------------------------------------------------------------------
// GCN clustering pipeline for MI455X (gfx1250, wave32).
// GEMMs: v_wmma_f32_16x16x32_f16 (f16 in, f32 accumulate).
// The two dominant wadj GEMMs stage their shared B strips through LDS with the
// Tensor Data Mover (tensor_load_to_lds + s_wait_tensorcnt double buffering).
// ---------------------------------------------------------------------------

typedef __attribute__((ext_vector_type(16))) _Float16     v16h;
typedef __attribute__((ext_vector_type(8)))  _Float16     v8h;
typedef __attribute__((ext_vector_type(8)))  float        v8f;
typedef __attribute__((ext_vector_type(4)))  unsigned int u32x4;
typedef __attribute__((ext_vector_type(8)))  int          i32x8;
typedef __attribute__((ext_vector_type(4)))  int          i32x4;

#define NPAD 8

// Load one 16-half WMMA fragment for a wave32 lane.
// ISA layout (16-bit A 16x32 / B 32x16): elems 0..7 = K (kbase..kbase+7),
// elems 8..15 = K (kbase+16..kbase+23), kbase = +8 for lanes 16..31.
__device__ __forceinline__ v16h load_frag(const _Float16* __restrict__ p) {
  v8h lo = *(const v8h*)p;        // 16B run
  v8h hi = *(const v8h*)(p + 16); // 16B run, +32B
  v16h r;
#pragma unroll
  for (int e = 0; e < 8; ++e) { r[e] = lo[e]; r[e + 8] = hi[e]; }
  return r;
}

// ---------------------------------------------------------------------------
// Direct-global WMMA GEMM (small K): register double-buffered K loop.
// C(MxN) = A(MxK f16 row-major) @ BT(NxK f16 row-major)^T, 16x64 strip/wave.
// ---------------------------------------------------------------------------
template <bool RELU, bool OUT16>
__global__ void gemm_wmma_f16(const _Float16* __restrict__ A, int lda,
                              const _Float16* __restrict__ BT, int ldb,
                              void* __restrict__ Cv, int ldc,
                              int tilesM, int stripsN, int K) {
  const int wave = threadIdx.x >> 5;
  const int lane = threadIdx.x & 31;
  const int gid  = blockIdx.x * (blockDim.x >> 5) + wave;
  if (gid >= tilesM * stripsN) return;
  const int m0   = (gid % tilesM) << 4;
  const int n0   = (gid / tilesM) << 6;
  const int l15  = lane & 15;
  const int koff = (lane >> 4) << 3;

  const _Float16* arow = A  + (size_t)(m0 + l15) * lda + koff;
  const _Float16* brow = BT + (size_t)(n0 + l15) * ldb + koff;

  v8f acc[4] = {};
  v16h a_cur = load_frag(arow);
  v16h b_cur[4];
#pragma unroll
  for (int t = 0; t < 4; ++t) b_cur[t] = load_frag(brow + (size_t)(t * 16) * ldb);

  for (int k = 32; k < K; k += 32) {
    v16h a_nxt = load_frag(arow + k);      // prefetch next fragments first so
    v16h b_nxt[4];                         // WMMAs overlap the VMEM latency
#pragma unroll
    for (int t = 0; t < 4; ++t) b_nxt[t] = load_frag(brow + (size_t)(t * 16) * ldb + k);
#pragma unroll
    for (int t = 0; t < 4; ++t)
      acc[t] = __builtin_amdgcn_wmma_f32_16x16x32_f16(
          false, a_cur, false, b_cur[t], (short)0, acc[t], false, false);
    a_cur = a_nxt;
#pragma unroll
    for (int t = 0; t < 4; ++t) b_cur[t] = b_nxt[t];
  }
#pragma unroll
  for (int t = 0; t < 4; ++t)
    acc[t] = __builtin_amdgcn_wmma_f32_16x16x32_f16(
        false, a_cur, false, b_cur[t], (short)0, acc[t], false, false);

  const int rbase = m0 + koff;
#pragma unroll
  for (int t = 0; t < 4; ++t) {
    const int col = n0 + t * 16 + l15;
#pragma unroll
    for (int r = 0; r < 8; ++r) {
      float v = acc[t][r];
      if (RELU) v = fmaxf(v, 0.f);
      const size_t o = (size_t)(rbase + r) * ldc + col;
      if (OUT16) ((_Float16*)Cv)[o] = (_Float16)v;
      else       ((float*)Cv)[o]    = v;
    }
  }
}

// ---------------------------------------------------------------------------
// TDM-staged WMMA GEMM for the dense wadj products (K = 3104).
// Per block: 8 m-tiles x one 64-col strip. Wave 0 DMAs the shared 64x32 f16
// B strip into LDS via tensor_load_to_lds (double buffered, TENSORcnt sync);
// D# pad feature inserts 16B per 64B row -> LDS row stride 40 halves
// (bank-conflict-free fragment reads).
// ---------------------------------------------------------------------------
#define BROW 40  // padded LDS row stride in halfwords (64B data + 16B pad)

__device__ __forceinline__ void tdm_load_b_strip(unsigned lds_off,
                                                 const _Float16* gsrc, int ldb) {
  unsigned long long ga = (unsigned long long)gsrc;
  u32x4 g0;
  g0[0] = 1u;                                 // count=1 (valid user descriptor)
  g0[1] = lds_off;                            // LDS byte address
  g0[2] = (unsigned)ga;                       // global_addr[31:0]
  g0[3] = (unsigned)(ga >> 32) | (2u << 30);  // global_addr[56:32] | type=2
  i32x8 g1;
  g1[0] = (1 << 16) | (1 << 20) | (3 << 22) | (3 << 25); // 2B elems; pad 4DW/16DW
  g1[1] = 32 << 16;  // tensor_dim0 = 32 (== tile_dim0, no OOB)
  g1[2] = 64 << 16;  // tensor_dim1 = 64 (== tile_dim1)
  g1[3] = 32 << 16;  // tile_dim0 = 32
  g1[4] = 64;        // tile_dim1 = 64, tile_dim2 = 0 (2D)
  g1[5] = ldb;       // tensor_dim0_stride (element units)
  g1[6] = 0;
  g1[7] = 0;
  i32x4 z4 = {0, 0, 0, 0};                    // groups 2/3 unused (<=2D tensor)
  i32x8 z8 = {0, 0, 0, 0, 0, 0, 0, 0};        // trailing group (6-arg toolchain)
  __builtin_amdgcn_tensor_load_to_lds(g0, g1, z4, z4, z8, 0);
}

template <bool RELU, bool OUT16>
__global__ void gemm_wmma_tdm(const _Float16* __restrict__ A, int lda,
                              const _Float16* __restrict__ BT, int ldb,
                              void* __restrict__ Cv, int ldc,
                              int tilesM, int K) {
  __shared__ _Float16 bbuf[2][64 * BROW];  // 2 x 5 KB double buffer
  const int wave = threadIdx.x >> 5;
  const int lane = threadIdx.x & 31;
  int mt = blockIdx.x * (blockDim.x >> 5) + wave;
  const bool active = (mt < tilesM);
  if (!active) mt = tilesM - 1;            // keep all waves in the barriers
  const int m0   = mt << 4;
  const int n0   = blockIdx.y << 6;
  const int l15  = lane & 15;
  const int koff = (lane >> 4) << 3;

  const _Float16* arow = A + (size_t)(m0 + l15) * lda + koff;
  const _Float16* bsrc = BT + (size_t)n0 * ldb;

  if (wave == 0)
    tdm_load_b_strip((unsigned)(unsigned long long)&bbuf[0][0], bsrc, ldb);

  v8f acc[4] = {};
  int nbuf = 0;
  for (int k = 0; k < K; k += 32) {
    if (wave == 0) {
      if (k + 32 < K) {
        tdm_load_b_strip((unsigned)(unsigned long long)&bbuf[nbuf ^ 1][0],
                         bsrc + k + 32, ldb);
        __builtin_amdgcn_s_wait_tensorcnt(1);  // current buffer complete
      } else {
        __builtin_amdgcn_s_wait_tensorcnt(0);
      }
    }
    v16h a = load_frag(arow + k);              // global A overlaps the barrier
    __syncthreads();                           // release bbuf[nbuf] to readers
#pragma unroll
    for (int t = 0; t < 4; ++t) {
      const _Float16* bp = &bbuf[nbuf][(t * 16 + l15) * BROW + koff];
      v8h lo = *(const v8h*)bp;
      v8h hi = *(const v8h*)(bp + 16);
      v16h b;
#pragma unroll
      for (int e = 0; e < 8; ++e) { b[e] = lo[e]; b[e + 8] = hi[e]; }
      acc[t] = __builtin_amdgcn_wmma_f32_16x16x32_f16(
          false, a, false, b, (short)0, acc[t], false, false);
    }
    __syncthreads();                           // all reads done before reuse
    nbuf ^= 1;
  }

  if (active) {
    const int rbase = m0 + koff;
#pragma unroll
    for (int t = 0; t < 4; ++t) {
      const int col = n0 + t * 16 + l15;
#pragma unroll
      for (int r = 0; r < 8; ++r) {
        float v = acc[t][r];
        if (RELU) v = fmaxf(v, 0.f);
        const size_t o = (size_t)(rbase + r) * ldc + col;
        if (OUT16) ((_Float16*)Cv)[o] = (_Float16)v;
        else       ((float*)Cv)[o]    = v;
      }
    }
  }
}

// --------------------------- small helper kernels --------------------------

__global__ void xpad_kernel(const float* __restrict__ ff, _Float16* __restrict__ xp,
                            int nrows, int d, int nsrc) {
  size_t idx = (size_t)blockIdx.x * blockDim.x + threadIdx.x;
  size_t tot = (size_t)nrows * d;
  if (idx >= tot) return;
  int i = (int)(idx / d), c = (int)(idx % d);
  int s = i - NPAD;
  s = s < 0 ? 0 : (s >= nsrc ? nsrc - 1 : s);
  xp[idx] = (_Float16)ff[(size_t)s * d + c];
}

// dst(N x dstLd f16) = transpose(src(K x N f32)); zero-pad k in [K,dstLd).
__global__ void transcvt_kernel(const float* __restrict__ src, _Float16* __restrict__ dst,
                                int K, int N, int dstLd) {
  size_t idx = (size_t)blockIdx.x * blockDim.x + threadIdx.x;
  size_t tot = (size_t)N * dstLd;
  if (idx >= tot) return;
  int n = (int)(idx / dstLd), k = (int)(idx % dstLd);
  float v = (k < K) ? src[(size_t)k * N + n] : 0.f;
  dst[idx] = (_Float16)v;
}

__global__ void tridiag_relu_kernel(const float* __restrict__ H, _Float16* __restrict__ oh,
                                    float* __restrict__ of, int n, int d) {
  int i = blockIdx.x;
  for (int c = threadIdx.x; c < d; c += blockDim.x) {
    float s = H[(size_t)i * d + c];
    if (i > 0)     s += H[(size_t)(i - 1) * d + c];
    if (i < n - 1) s += H[(size_t)(i + 1) * d + c];
    s = fmaxf(s, 0.f);
    oh[(size_t)i * d + c] = (_Float16)s;
    if (of) of[(size_t)i * d + c] = s;
  }
}

__global__ void diffnorm_kernel(const float* __restrict__ f1, float* __restrict__ dvec,
                                int nd, int d) {
  int wave = threadIdx.x >> 5, lane = threadIdx.x & 31;
  int i = blockIdx.x * (blockDim.x >> 5) + wave;
  if (i >= nd) return;
  float s = 0.f;
  for (int c = lane; c < d; c += 32) {
    float df = f1[(size_t)(i + 1) * d + c] - f1[(size_t)i * d + c];
    s += df * df;
  }
  for (int off = 16; off > 0; off >>= 1) s += __shfl_down(s, off, 32);
  if (lane == 0) dvec[i] = sqrtf(s);
}

__global__ void scan_kernel(const float* __restrict__ d, float* __restrict__ accs,
                            int nd, float invN) {
  __shared__ float part[1024];
  const int t = threadIdx.x;
  float loc[4];
  float s = 0.f;
#pragma unroll
  for (int c = 0; c < 4; ++c) {
    int i = t * 4 + c;
    float v = (i < nd) ? d[i] : 0.f;
    loc[c] = v; s += v;
  }
  part[t] = s;
  __syncthreads();
  for (int off = 1; off < 1024; off <<= 1) {
    float v = part[t];
    float w = (t >= off) ? part[t - off] : 0.f;
    __syncthreads();
    part[t] = v + w;
    __syncthreads();
  }
  float total = part[1023];
  float sc = invN * (float)nd / total;       // 1/(mean*n)
  float run = (t > 0) ? part[t - 1] : 0.f;
#pragma unroll
  for (int c = 0; c < 4; ++c) {
    int i = t * 4 + c;
    run += loc[c];
    if (i < nd) accs[i + 1] = run * sc;
  }
  if (t == 0) accs[0] = 0.f;
}

__global__ void wadj_kernel(const float* __restrict__ accs, _Float16* __restrict__ w,
                            int n, int ld) {
  size_t idx = (size_t)blockIdx.x * blockDim.x + threadIdx.x;
  size_t tot = (size_t)n * ld;
  if (idx >= tot) return;
  int i = (int)(idx / ld), j = (int)(idx % ld);
  float v = 0.f;
  if (j < n) v = __expf(-fabsf(accs[i] - accs[j]));
  w[idx] = (_Float16)v;
}

__global__ void score_kernel(const float* __restrict__ F2, const float* __restrict__ fcw,
                             const float* __restrict__ fcb, float* __restrict__ scores,
                             int nout, int d) {
  int wave = threadIdx.x >> 5, lane = threadIdx.x & 31;
  int r = blockIdx.x * (blockDim.x >> 5) + wave;
  if (r >= nout) return;
  const float* row = F2 + (size_t)(r + NPAD) * d;
  float s = 0.f;
  for (int c = lane; c < d; c += 32) s += row[c] * fcw[c];
  for (int off = 16; off > 0; off >>= 1) s += __shfl_down(s, off, 32);
  if (lane == 0) scores[r] = s + fcb[0];
}

__global__ void segmean_kernel(const float* __restrict__ scores, const int* __restrict__ seg,
                               float* __restrict__ out, int nseg, int sl) {
  int wave = threadIdx.x >> 5, lane = threadIdx.x & 31;
  int s = blockIdx.x * (blockDim.x >> 5) + wave;
  if (s >= nseg) return;
  float a = 0.f;
  for (int k = lane; k < sl; k += 32) a += scores[seg[(size_t)s * sl + k]];
  for (int off = 16; off > 0; off >>= 1) a += __shfl_down(a, off, 32);
  if (lane == 0) out[s] = a / (float)sl;
}

// ------------------------------- launcher ----------------------------------

extern "C" void kernel_launch(void* const* d_in, const int* in_sizes, int n_in,
                              void* d_out, int out_size, void* d_ws, size_t ws_size,
                              hipStream_t stream) {
  (void)in_sizes; (void)n_in; (void)out_size; (void)ws_size;
  const float* ff  = (const float*)d_in[0];
  const int*   seg = (const int*)  d_in[1];
  const float* w10 = (const float*)d_in[2];  // 512x512
  const float* w11 = (const float*)d_in[3];  // 512x256
  const float* w20 = (const float*)d_in[4];  // 256x256
  const float* w21 = (const float*)d_in[5];  // 256x128
  const float* fcw = (const float*)d_in[6];  // 128
  const float* fcb = (const float*)d_in[7];  // 1

  const int N0 = 3072, NR = N0 + 2 * NPAD /*3088*/, KP = 3104 /*NR -> %32*/;

  char* base = (char*)d_ws; size_t off = 0;
  auto alloc = [&](size_t bytes) -> void* {
    void* r = base + off;
    off += (bytes + 255) & ~(size_t)255;
    return r;
  };
  _Float16* w10T  = (_Float16*)alloc((size_t)512 * 512 * 2);
  _Float16* w11T  = (_Float16*)alloc((size_t)256 * 512 * 2);
  _Float16* w20T  = (_Float16*)alloc((size_t)256 * 256 * 2);
  _Float16* w21T  = (_Float16*)alloc((size_t)128 * 256 * 2);
  _Float16* xpadU = (_Float16*)alloc((size_t)NR * 512 * 2);  // xpad -> later U_h
  float*    big   = (float*)   alloc((size_t)NR * 512 * 4);  // H -> H2 -> S1 -> S2 -> F2
  _Float16* t1s   = (_Float16*)alloc((size_t)NR * 512 * 2);  // T1_h -> S1T -> S2T
  float*    f1f   = (float*)   alloc((size_t)NR * 256 * 4);
  _Float16* f1h   = (_Float16*)alloc((size_t)NR * 256 * 2);
  float*    dvec  = (float*)   alloc((size_t)(NR - 1) * 4);
  float*    accs  = (float*)   alloc((size_t)NR * 4);
  _Float16* wadj  = (_Float16*)alloc((size_t)NR * KP * 2);   // 19 MB, L2-resident
  float*    scores= (float*)   alloc((size_t)N0 * 4);

  auto cdiv = [](size_t a, size_t b) { return (int)((a + b - 1) / b); };
  const int tilesM = NR / 16;  // 193

  transcvt_kernel<<<cdiv((size_t)512 * 512, 256), 256, 0, stream>>>(w10, w10T, 512, 512, 512);
  transcvt_kernel<<<cdiv((size_t)256 * 512, 256), 256, 0, stream>>>(w11, w11T, 512, 256, 512);
  transcvt_kernel<<<cdiv((size_t)256 * 256, 256), 256, 0, stream>>>(w20, w20T, 256, 256, 256);
  transcvt_kernel<<<cdiv((size_t)128 * 256, 256), 256, 0, stream>>>(w21, w21T, 256, 128, 256);
  xpad_kernel<<<cdiv((size_t)NR * 512, 256), 256, 0, stream>>>(ff, xpadU, NR, 512, N0);

  // GCN layer 1.
  gemm_wmma_f16<false, false><<<cdiv((size_t)tilesM * 8, 8), 256, 0, stream>>>(
      xpadU, 512, w10T, 512, big, 512, tilesM, 8, 512);                // H = Xp @ W1_0
  tridiag_relu_kernel<<<NR, 256, 0, stream>>>(big, t1s, nullptr, NR, 512);
  gemm_wmma_f16<false, false><<<cdiv((size_t)tilesM * 4, 8), 256, 0, stream>>>(
      t1s, 512, w11T, 512, big, 256, tilesM, 4, 512);                  // H2 = T1 @ W1_1
  tridiag_relu_kernel<<<NR, 256, 0, stream>>>(big, f1h, f1f, NR, 256); // f1

  // Weighted adjacency.
  diffnorm_kernel<<<cdiv((size_t)(NR - 1), 8), 256, 0, stream>>>(f1f, dvec, NR - 1, 256);
  scan_kernel<<<1, 1024, 0, stream>>>(dvec, accs, NR - 1, 1.0f / (float)NR);
  wadj_kernel<<<cdiv((size_t)NR * KP, 256), 256, 0, stream>>>(accs, wadj, NR, KP);

  // GCN layer 2 (dense wadj GEMMs: TDM-staged B strips).
  gemm_wmma_f16<false, false><<<cdiv((size_t)tilesM * 4, 8), 256, 0, stream>>>(
      f1h, 256, w20T, 256, big, 256, tilesM, 4, 256);                  // S1 = f1 @ W2_0
  transcvt_kernel<<<cdiv((size_t)256 * KP, 256), 256, 0, stream>>>(big, t1s, NR, 256, KP);
  {
    dim3 g(cdiv((size_t)tilesM, 8), 4);
    gemm_wmma_tdm<true, true><<<g, 256, 0, stream>>>(
        wadj, KP, t1s, KP, xpadU, 256, tilesM, KP);                    // U = relu(wadj@S1)
  }
  gemm_wmma_f16<false, false><<<cdiv((size_t)tilesM * 2, 8), 256, 0, stream>>>(
      xpadU, 256, w21T, 256, big, 128, tilesM, 2, 256);                // S2 = U @ W2_1
  transcvt_kernel<<<cdiv((size_t)128 * KP, 256), 256, 0, stream>>>(big, t1s, NR, 128, KP);
  {
    dim3 g(cdiv((size_t)tilesM, 8), 2);
    gemm_wmma_tdm<true, false><<<g, 256, 0, stream>>>(
        wadj, KP, t1s, KP, big, 128, tilesM, KP);                      // F2 = relu(wadj@S2)
  }

  // Head.
  score_kernel<<<cdiv((size_t)N0, 8), 256, 0, stream>>>(big, fcw, fcb, scores, N0, 128);
  segmean_kernel<<<cdiv((size_t)256, 8), 256, 0, stream>>>(scores, seg, (float*)d_out, 256, 64);
}
